// SharedSinkhorn_87960930222187
// MI455X (gfx1250) — matrix operand order repaired and checked by reading, not compiled, not run on hardware
//
#include <hip/hip_runtime.h>
#include <hip/hip_bf16.h>
#include <math.h>

// ---------- problem constants ----------
#define BB   32
#define CC   768
#define HW   1024
#define DD   384
#define KK   64
#define INV_EPS 20.0f
#define SINK_ITERS 10
#define NORM_CONST (-6.9920964274158895f)   // -log(64+1024)

typedef __attribute__((ext_vector_type(16))) _Float16 v16h;
typedef __attribute__((ext_vector_type(8)))  _Float16 v8h;
typedef __attribute__((ext_vector_type(8)))  float    v8f;

union V16 { v16h v; v8h v8[2]; _Float16 h[16]; };
union V8H { v8h v; _Float16 h[8]; };

// ======================================================================
// K0: one-time f32 -> f16 conversion of conv_w (1.2MB -> 0.6MB, L2 resident)
// ======================================================================
__global__ void __launch_bounds__(256) k_cvt_f16(const float* __restrict__ src,
                                                 _Float16* __restrict__ dst,
                                                 int nelem) {
  int i = blockIdx.x * 256 + threadIdx.x;
  if (i < nelem) dst[i] = (_Float16)src[i];
}

// ======================================================================
// K1 (fused): x_proj[b][n][d] = sum_c x[b][c][n]*W[d][c] + bias[d]  (f32 out)
//             xn[b][n][d]     = x_proj / max(||x_proj||_2, 1e-12)   (f16 out)
// WMMA: A = Wh tile (M=d,K=c, rows contiguous f16 -> 2x b128 global),
//       B = x tile  (K=c,N=n, staged to LDS [nt][n][k], double-buffered).
// Block 256 = 8 waves; wave owns 3 d-tiles x 2 n-tiles (6 accumulators).
// Staging: one coalesced float4 per thread per k-step (k=t>>3, nq=(t&7)*4).
// WG holds full D for its 32 tokens -> in-WG deterministic L2-norm.
// ======================================================================
__global__ void __launch_bounds__(256) k_proj(const float* __restrict__ x,
                                              const _Float16* __restrict__ Wh,
                                              const float* __restrict__ bias,
                                              float* __restrict__ xproj,
                                              _Float16* __restrict__ xn) {
  __shared__ _Float16 ldsB[2][1024];  // double-buffered [nt 2][n 16][k 32]
  __shared__ float part[8][32];       // per-wave partial |x|^2 per token
  __shared__ float norm2[32];
  const int tid  = threadIdx.x;
  const int lane = tid & 31;
  const int wv   = tid >> 5;
  const int hi   = lane >> 4;         // 0: lanes 0-15, 1: lanes 16-31
  const int l15  = lane & 15;
  const int n0   = blockIdx.x * 32;
  const int b    = blockIdx.y;
  const size_t xb = (size_t)b * CC * HW;
  const int dtb  = wv * 48;           // wave's d base (3 tiles of 16)

  // staging coordinates: one float4 per thread per k-step
  const int sk = tid >> 3;            // c-row within block (0..31)
  const int nq = (tid & 7) * 4;       // n-quad (0,4,...,28)
  const int sbase = (nq >> 4) * 512 + (nq & 15) * 32 + sk;

  auto stage = [&](int c0, int buf) {
    float4 f = *reinterpret_cast<const float4*>(
        &x[xb + (size_t)(c0 + sk) * HW + n0 + nq]);
    _Float16* p = &ldsB[buf][0];
    p[sbase]      = (_Float16)f.x;
    p[sbase + 32] = (_Float16)f.y;
    p[sbase + 64] = (_Float16)f.z;
    p[sbase + 96] = (_Float16)f.w;
  };

  v8f acc[2][3] = {};
  stage(0, 0);

  for (int s = 0; s < CC / 32; ++s) {
    const int c0  = s * 32;
    const int buf = s & 1;
    __syncthreads();                  // stage(s) complete; reads of buf^1 done

    V16 bv[2];
#pragma unroll
    for (int nt = 0; nt < 2; ++nt) {
      const int bbase = nt * 512 + l15 * 32 + hi * 8;
      bv[nt].v8[0] = *reinterpret_cast<const v8h*>(&ldsB[buf][bbase]);
      bv[nt].v8[1] = *reinterpret_cast<const v8h*>(&ldsB[buf][bbase + 16]);
    }

    if (s + 1 < CC / 32) stage(c0 + 32, buf ^ 1);   // overlap with WMMAs
    if (s + 2 < CC / 32)                            // gfx1250 global_prefetch_b8
      __builtin_prefetch(&x[xb + (size_t)(c0 + 64 + sk) * HW + n0 + nq], 0, 3);

    const int clo = c0 + hi * 8;
#pragma unroll
    for (int j = 0; j < 3; ++j) {
      const int d = dtb + j * 16 + l15;          // A row per lane (f16, b128 x2)
      const _Float16* ap = Wh + (size_t)d * CC + clo;
      V16 av;
      av.v8[0] = *reinterpret_cast<const v8h*>(ap);
      av.v8[1] = *reinterpret_cast<const v8h*>(ap + 16);
#pragma unroll
      for (int nt = 0; nt < 2; ++nt)
        acc[nt][j] = __builtin_amdgcn_wmma_f32_16x16x32_f16(
            false, av.v, false, bv[nt].v, (short)0, acc[nt][j], false, false);
    }
  }

  // bias add (bias depends only on d -> load once per d-tile)
#pragma unroll
  for (int j = 0; j < 3; ++j) {
    const int dstart = dtb + j * 16 + hi * 8;
    float4 b0 = *reinterpret_cast<const float4*>(bias + dstart);
    float4 b1 = *reinterpret_cast<const float4*>(bias + dstart + 4);
#pragma unroll
    for (int nt = 0; nt < 2; ++nt) {
      acc[nt][j][0] += b0.x; acc[nt][j][1] += b0.y;
      acc[nt][j][2] += b0.z; acc[nt][j][3] += b0.w;
      acc[nt][j][4] += b1.x; acc[nt][j][5] += b1.y;
      acc[nt][j][6] += b1.z; acc[nt][j][7] += b1.w;
    }
  }

  // deterministic norm reduction: lane partial -> pair-combine -> LDS tree
#pragma unroll
  for (int nt = 0; nt < 2; ++nt) {
    float ss = 0.f;
#pragma unroll
    for (int j = 0; j < 3; ++j)
#pragma unroll
      for (int r = 0; r < 8; ++r) ss += acc[nt][j][r] * acc[nt][j][r];
    ss += __shfl_xor(ss, 16);                    // combine d-halves (hi/lo)
    if (hi == 0) part[wv][nt * 16 + l15] = ss;   // wave partial for token
    __syncthreads();
    if (tid < 32) {
      float s = 0.f;
#pragma unroll
      for (int w = 0; w < 8; ++w) s += part[w][tid];   // fixed order
      norm2[tid] = s;
    }
    __syncthreads();
  }

  // store x_proj (f32) and xn (f16)
#pragma unroll
  for (int nt = 0; nt < 2; ++nt) {
    const int tok_l = nt * 16 + l15;
    const int n     = n0 + tok_l;
    const float inv = 1.0f / fmaxf(sqrtf(norm2[tok_l]), 1e-12f);
#pragma unroll
    for (int j = 0; j < 3; ++j) {
      const int dstart = dtb + j * 16 + hi * 8;
      float4 o0, o1;
      o0.x = acc[nt][j][0]; o0.y = acc[nt][j][1];
      o0.z = acc[nt][j][2]; o0.w = acc[nt][j][3];
      o1.x = acc[nt][j][4]; o1.y = acc[nt][j][5];
      o1.z = acc[nt][j][6]; o1.w = acc[nt][j][7];
      float* op = xproj + ((size_t)b * HW + n) * DD + dstart;
      *reinterpret_cast<float4*>(op)     = o0;
      *reinterpret_cast<float4*>(op + 4) = o1;
      V8H q;
#pragma unroll
      for (int r = 0; r < 8; ++r) q.h[r] = (_Float16)(acc[nt][j][r] * inv);
      *reinterpret_cast<v8h*>(xn + ((size_t)b * HW + n) * DD + dstart) = q.v;
    }
  }
}

// ======================================================================
// K2: per-token L2 normalize 384-dim rows, emit f16 (used for codebook v).
// ======================================================================
__global__ void __launch_bounds__(256) k_l2norm(const float* __restrict__ src,
                                                _Float16* __restrict__ dst,
                                                int ntok) {
  const int lane = threadIdx.x & 31;
  const int tok  = blockIdx.x * 8 + (threadIdx.x >> 5);
  if (tok >= ntok) return;
  const float* p = src + (size_t)tok * DD;
  float vals[12];
  float s = 0.f;
#pragma unroll
  for (int j = 0; j < 12; ++j) { float t = p[lane + 32 * j]; vals[j] = t; s += t * t; }
#pragma unroll
  for (int off = 16; off; off >>= 1) s += __shfl_xor(s, off);
  const float inv = 1.0f / fmaxf(sqrtf(s), 1e-12f);
  _Float16* q = dst + (size_t)tok * DD;
#pragma unroll
  for (int j = 0; j < 12; ++j) q[lane + 32 * j] = (_Float16)(vals[j] * inv);
}

// ======================================================================
// K3: Z[b][k][n] = dot(clusters[k], xn[b][n]) / eps    (f32, Sinkhorn input)
// A = clusters (rows contiguous), B = xn (columns contiguous since [n][d]).
// ======================================================================
__global__ void __launch_bounds__(256) k_scores(const _Float16* __restrict__ xn,
                                                const _Float16* __restrict__ clus,
                                                float* __restrict__ Z) {
  const int lane = threadIdx.x & 31;
  const int wv   = threadIdx.x >> 5;
  const int hi   = lane >> 4;
  const int l15  = lane & 15;
  const int b  = blockIdx.z, kt = blockIdx.y;
  const int n0 = blockIdx.x * 128 + wv * 16;
  const int k  = kt * 16 + l15;
  const int n  = n0 + l15;

  v8f acc = {};
  for (int d0 = 0; d0 < DD; d0 += 32) {
    V16 av, bv;
    const _Float16* ap = clus + (size_t)k * DD + d0 + hi * 8;
    av.v8[0] = *reinterpret_cast<const v8h*>(ap);
    av.v8[1] = *reinterpret_cast<const v8h*>(ap + 16);
    const _Float16* bp = xn + ((size_t)b * HW + n) * DD + d0 + hi * 8;
    bv.v8[0] = *reinterpret_cast<const v8h*>(bp);
    bv.v8[1] = *reinterpret_cast<const v8h*>(bp + 16);
    acc = __builtin_amdgcn_wmma_f32_16x16x32_f16(
        false, av.v, false, bv.v, (short)0, acc, false, false);
  }
  float* Zb = Z + (size_t)b * KK * HW;
#pragma unroll
  for (int r = 0; r < 8; ++r) {
    const int kk = kt * 16 + r + hi * 8;
    Zb[(size_t)kk * HW + n] = acc[r] * INV_EPS;
  }
}

// ======================================================================
// K4: log-Sinkhorn, one 1024-thread workgroup per batch image.
// Z stays in global (L2-resident, 256KB/batch); u[64], v[1024] in LDS.
// Fast hardware transcendentals (v_exp_f32/v_log_f32) shorten the
// latency-critical online-logsumexp dependency chain.
// ======================================================================
__global__ void __launch_bounds__(1024) k_sinkhorn(const float* __restrict__ Z,
                                                   _Float16* __restrict__ wgt) {
  __shared__ float u_s[KK];
  __shared__ float v_s[HW];
  const int t    = threadIdx.x;
  const int lane = t & 31;
  const int wv   = t >> 5;
  const int b    = blockIdx.x;
  const float* Zb = Z + (size_t)b * KK * HW;
  const float NEG_INF = -__builtin_inff();

  v_s[t] = 0.f;
  __syncthreads();

  for (int it = 0; it < SINK_ITERS; ++it) {
    // u-step: u[k] = NORM - logsumexp_n(Z[k][n] + v[n]); wave wv -> rows wv, wv+32
#pragma unroll
    for (int rr = 0; rr < 2; ++rr) {
      const int k = wv + rr * 32;
      const float* zr = Zb + (size_t)k * HW;
      float vals[32];
      float m = NEG_INF;
#pragma unroll
      for (int j = 0; j < 32; ++j) {
        const int n = j * 32 + lane;
        float z = zr[n] + v_s[n];
        vals[j] = z;
        m = fmaxf(m, z);
      }
#pragma unroll
      for (int off = 16; off; off >>= 1) m = fmaxf(m, __shfl_xor(m, off));
      float s = 0.f;
#pragma unroll
      for (int j = 0; j < 32; ++j) s += __expf(vals[j] - m);
#pragma unroll
      for (int off = 16; off; off >>= 1) s += __shfl_xor(s, off);
      if (lane == 0) u_s[k] = NORM_CONST - (m + __logf(s));
    }
    __syncthreads();

    // v-step: thread t == column n; online logsumexp over k (coalesced rows)
    {
      float m = NEG_INF, s = 0.f;
      for (int k = 0; k < KK; ++k) {
        float a  = Zb[(size_t)k * HW + t] + u_s[k];
        float nm = fmaxf(m, a);
        s = s * __expf(m - nm) + __expf(a - nm);
        m = nm;
      }
      v_s[t] = NORM_CONST - (m + __logf(s));
    }
    __syncthreads();
  }

  // weights[b][k][n] = exp(Z + u + v - NORM), stored k-major f16 for GEMM-3 A
#pragma unroll 4
  for (int k = 0; k < KK; ++k) {
    float z = Zb[(size_t)k * HW + t] + u_s[k] + v_s[t] - NORM_CONST;
    wgt[(size_t)b * KK * HW + (size_t)k * HW + t] = (_Float16)__expf(z);
  }
}

// ======================================================================
// K5: v_tilde[b][k][d] = sum_n wgt[b][k][n] * xn[b][n][d]
// A = wgt (k rows contiguous in n), B = xn d-columns staged via LDS.
// ======================================================================
__global__ void __launch_bounds__(256) k_vtilde(const _Float16* __restrict__ xn,
                                                const _Float16* __restrict__ wgt,
                                                float* __restrict__ out) {
  __shared__ _Float16 lds[8 * 512];              // per-wave [d 16][n 32]
  const int lane = threadIdx.x & 31;
  const int wv   = threadIdx.x >> 5;
  const int hi   = lane >> 4;
  const int l15  = lane & 15;
  const int b    = blockIdx.y;
  const int d0   = blockIdx.x * 128 + wv * 16;
  _Float16* myB  = lds + wv * 512;

  v8f acc[4] = {};

  for (int n0 = 0; n0 < HW; n0 += 32) {
    union { v8h v[2]; _Float16 h[16]; } row;
    const _Float16* rp = xn + ((size_t)b * HW + n0 + lane) * DD + d0;
    row.v[0] = *reinterpret_cast<const v8h*>(rp);
    row.v[1] = *reinterpret_cast<const v8h*>(rp + 8);
    __syncthreads();                             // protect prior-iter LDS reads
#pragma unroll
    for (int d = 0; d < 16; ++d) myB[d * 32 + lane] = row.h[d];
    __syncthreads();

    V16 bv;
    const int bbase = l15 * 32 + hi * 8;
    bv.v8[0] = *reinterpret_cast<const v8h*>(&myB[bbase]);
    bv.v8[1] = *reinterpret_cast<const v8h*>(&myB[bbase + 16]);

    const int nb = n0 + hi * 8;
#pragma unroll
    for (int kt = 0; kt < 4; ++kt) {
      const int kk = kt * 16 + l15;
      const _Float16* ap = wgt + ((size_t)b * KK + kk) * HW + nb;
      V16 av;
      av.v8[0] = *reinterpret_cast<const v8h*>(ap);
      av.v8[1] = *reinterpret_cast<const v8h*>(ap + 16);
      acc[kt] = __builtin_amdgcn_wmma_f32_16x16x32_f16(
          false, av.v, false, bv.v, (short)0, acc[kt], false, false);
    }
  }

  const int d = d0 + l15;
#pragma unroll
  for (int kt = 0; kt < 4; ++kt) {
#pragma unroll
    for (int r = 0; r < 8; ++r) {
      const int kk = kt * 16 + r + hi * 8;
      out[((size_t)b * KK + kk) * DD + d] = acc[kt][r];
    }
  }
}

// ======================================================================
extern "C" void kernel_launch(void* const* d_in, const int* in_sizes, int n_in,
                              void* d_out, int out_size, void* d_ws, size_t ws_size,
                              hipStream_t stream) {
  const float* x      = (const float*)d_in[0];   // [32,768,32,32]
  const float* conv_w = (const float*)d_in[1];   // [384,768]
  const float* conv_b = (const float*)d_in[2];   // [384]
  const float* v      = (const float*)d_in[3];   // [64,384]

  float* vtilde_out = (float*)d_out;                         // [32,64,384]
  float* xproj_out  = (float*)d_out + (size_t)BB * KK * DD;  // [32,1024,384]

  // workspace layout (all 16B aligned)
  _Float16* xn_h   = (_Float16*)d_ws;                        // 32*1024*384 halves
  _Float16* clus_h = xn_h + (size_t)BB * HW * DD;            // 64*384 halves
  _Float16* w_h    = clus_h + (size_t)KK * DD;               // 384*768 halves
  float*    Z      = (float*)(w_h + (size_t)DD * CC);        // 32*64*1024 f32
  _Float16* wgt_h  = (_Float16*)(Z + (size_t)BB * KK * HW);  // 32*64*1024 halves

  // 0) conv_w -> f16 once (L2-resident afterwards)
  k_cvt_f16<<<dim3((DD * CC + 255) / 256), 256, 0, stream>>>(conv_w, w_h, DD * CC);
  // 1) fused 1x1-conv GEMM + bias + token L2-norm -> x_proj (f32) and xn (f16)
  k_proj<<<dim3(HW / 32, BB), 256, 0, stream>>>(x, w_h, conv_b, xproj_out, xn_h);
  // 2) codebook L2-norm -> f16
  k_l2norm<<<dim3(KK / 8), 256, 0, stream>>>(v, clus_h, KK);
  // 3) scores / eps -> Z[b][k][n]
  k_scores<<<dim3(8, 4, BB), 256, 0, stream>>>(xn_h, clus_h, Z);
  // 4) Sinkhorn (10 iters) -> weights f16 [b][k][n]
  k_sinkhorn<<<dim3(BB), 1024, 0, stream>>>(Z, wgt_h);
  // 5) v_tilde[b][k][d]
  k_vtilde<<<dim3(3, BB), 256, 0, stream>>>(xn_h, wgt_h, vtilde_out);
}
